// MyConv2d_80874234183957
// MI455X (gfx1250) — matrix operand order repaired and checked by reading, not compiled, not run on hardware
//
#include <hip/hip_runtime.h>

// CDNA5 / gfx1250 implicit-GEMM 3x3 conv (pad=1, stride=1), fp32 in/out,
// computed with bf16x3-split V_WMMA_F32_16X16X32_BF16 (f32 accumulate).
//
// Fast path (needs ~35 MB workspace):
//   prologue kernels split f32 -> bf16(hi)+bf16(lo) planes once:
//     weights -> [9][Cout=256][Cin=128]   (A tiles: K=cin contiguous)
//     x       -> NHWC [16][64][64][128]   (B tiles: K=cin contiguous)
//   main kernel: 9 shifted GEMMs (kh,kw), K=128 each, BK=32 -> 36 K-steps.
//   Staging is pure 16B global->LDS async-DMA copies (ASYNCcnt), double-
//   buffered so copies of tile k+1 overlap the 6 WMMAs of tile k.
// Fallback path (small ws): round-1 self-contained kernel.

typedef __attribute__((ext_vector_type(16))) __bf16 bf16x16;
typedef __attribute__((ext_vector_type(8)))  float  f32x8;
typedef unsigned short ushort_t;
typedef int v4i __attribute__((vector_size(16)));

union Frag {
    uint4   q[2];
    bf16x16 v;
};

#define WMMA_BF16(A, B, C) \
    __builtin_amdgcn_wmma_f32_16x16x32_bf16(false, (A), false, (B), (short)0, (C), false, false)

#if __has_builtin(__builtin_amdgcn_global_load_async_to_lds_b128)
#define HAVE_ASYNC_LDS 1
#else
#define HAVE_ASYNC_LDS 0
#endif

// hi = truncate-to-bf16(x) (exact), lo = RNE-bf16(x - hi). hi*hi+hi*lo+lo*hi
// recovers ~16-17 mantissa bits -> near-fp32 GEMM accuracy.
__device__ __forceinline__ void split_bf16x2(float x, ushort_t& hi, ushort_t& lo) {
    unsigned u  = __builtin_bit_cast(unsigned, x);
    ushort_t h  = (ushort_t)(u >> 16);
    float hf    = __builtin_bit_cast(float, (unsigned)h << 16);
    float l     = x - hf;
    __bf16 lb   = (__bf16)l;
    hi = h;
    lo = __builtin_bit_cast(ushort_t, lb);
}

__device__ __forceinline__ uint4 pack8(const ushort_t* s) {
    uint4 q;
    q.x = (unsigned)s[0] | ((unsigned)s[1] << 16);
    q.y = (unsigned)s[2] | ((unsigned)s[3] << 16);
    q.z = (unsigned)s[4] | ((unsigned)s[5] << 16);
    q.w = (unsigned)s[6] | ((unsigned)s[7] << 16);
    return q;
}

// 16-byte global -> LDS copy; async DMA path (ASYNCcnt) when the builtin is
// available, synchronous b128 load + ds_store otherwise.
__device__ __forceinline__ void copy16_g2l(ushort_t* lds, const ushort_t* g) {
#if HAVE_ASYNC_LDS
    __builtin_amdgcn_global_load_async_to_lds_b128(
        (__attribute__((address_space(1))) v4i*)(v4i*)(ushort_t*)g,
        (__attribute__((address_space(3))) v4i*)(v4i*)lds,
        0, 0);
#else
    *(uint4*)lds = *(const uint4*)g;
#endif
}

__device__ __forceinline__ void wait_async0() {
#if HAVE_ASYNC_LDS
#if __has_builtin(__builtin_amdgcn_s_wait_asynccnt)
    __builtin_amdgcn_s_wait_asynccnt(0);
#else
    asm volatile("s_wait_asynccnt 0x0" ::: "memory");
#endif
#endif
}

// ---------------------------------------------------------------------------
// Prologue 1: weight [256][128][3][3] f32 -> whi/wlo [9][256][128] bf16
// ---------------------------------------------------------------------------
__global__ __launch_bounds__(256)
void prep_weight(const float* __restrict__ w, ushort_t* __restrict__ whi,
                 ushort_t* __restrict__ wlo) {
    int i = blockIdx.x * 256 + threadIdx.x;          // 294912 elements
    int cout = i / 1152;
    int rem  = i - cout * 1152;
    int cin  = rem / 9;
    int t    = rem - cin * 9;                        // kh*3+kw
    ushort_t h, l;
    split_bf16x2(w[i], h, l);
    int o = (t * 256 + cout) * 128 + cin;
    whi[o] = h;
    wlo[o] = l;
}

// ---------------------------------------------------------------------------
// Prologue 2: x NCHW [16][128][64][64] f32 -> xhi/xlo NHWC [16][64][64][128]
// ---------------------------------------------------------------------------
__global__ __launch_bounds__(256)
void prep_x(const float* __restrict__ x, ushort_t* __restrict__ xhi,
            ushort_t* __restrict__ xlo) {
    int tid = blockIdx.x * 256 + threadIdx.x;        // 2,097,152 threads
    int pix = tid >> 5;                              // b*4096 + h*64 + w
    int c4  = (tid & 31) * 4;                        // cin base (4 per thread)
    int b   = pix >> 12;
    int hw  = pix & 4095;
    ushort_t h4[4], l4[4];
#pragma unroll
    for (int j = 0; j < 4; ++j) {
        float v = x[(b * 128 + c4 + j) * 4096 + hw];
        split_bf16x2(v, h4[j], l4[j]);
    }
    uint2 ph, pl;
    ph.x = (unsigned)h4[0] | ((unsigned)h4[1] << 16);
    ph.y = (unsigned)h4[2] | ((unsigned)h4[3] << 16);
    pl.x = (unsigned)l4[0] | ((unsigned)l4[1] << 16);
    pl.y = (unsigned)l4[2] | ((unsigned)l4[3] << 16);
    int o = pix * 128 + c4;
    *(uint2*)&xhi[o] = ph;
    *(uint2*)&xlo[o] = pl;
}

// ---------------------------------------------------------------------------
// Main kernel: 9 shifted GEMMs, double-buffered async staging.
// Block tile BM=64(Cout) x BN=64(one image row) x BK=32(cin); 8 wave32.
// wave w: M-subtile w>>1, N-subtiles {2*(w&1), 2*(w&1)+1}.
// ---------------------------------------------------------------------------
__global__ __launch_bounds__(256)
void conv3x3_wmma_fast(const ushort_t* __restrict__ whi,
                       const ushort_t* __restrict__ wlo,
                       const ushort_t* __restrict__ xhi,
                       const ushort_t* __restrict__ xlo,
                       float* __restrict__ out) {
    __shared__ ushort_t Ahi[2][64 * 32];
    __shared__ ushort_t Alo[2][64 * 32];
    __shared__ ushort_t Bhi[2][64 * 32];   // [n][k], K contiguous
    __shared__ ushort_t Blo[2][64 * 32];

    const int tid = threadIdx.x;
    const int bx  = blockIdx.x;            // b*64 + h
    const int m0  = blockIdx.y * 64;       // Cout block
    const int b   = bx >> 6;
    const int h   = bx & 63;

    // Staging mapping: 64 rows x 32 k, 16B (8 bf16) per thread per plane.
    const int row = tid >> 2;              // 0..63 (A: cout row, B: pixel w)
    const int kg  = (tid & 3) * 8;         // cin offset group

    // Per-thread invariant address parts.
    const int arow  = (m0 + row) * 128 + kg;      // into whi[khw*32768 + ...]
    const int brow  = row * 128 + kg;             // win*128 folded per phase
    const int ldsA  = row * 32 + kg;
    const int ldsB  = row * 32 + kg;
    const int pixB  = b * 4096;                   // + hin*64 (+win)

    // Fragment mapping.
    const int lane = tid & 31;
    const int wave = tid >> 5;
    const int mt   = wave >> 1;
    const int np   = wave & 1;
    const int l15  = lane & 15;
    const int lh   = lane >> 4;
    const int aoff0 = (mt * 16 + l15) * 32 + lh * 16;  // A: K0-15 / K16-31
    const int aoff1 = aoff0 + 8;
    const int b0off = (np * 32 + l15) * 32 + lh * 16;  // B: K0-15 / K16-31
    const int b1off = b0off + 16 * 32;

    f32x8 acc0 = {};
    f32x8 acc1 = {};

    // ---- stage K-step `ks` (0..35) into buffer `buf`
    auto stage = [&](int ks, int buf) {
        const int khw = ks >> 2;           // 0..8 = kh*3+kw
        const int cb  = (ks & 3) * 32;     // cin block base
        const int kh  = khw / 3;
        const int kw  = khw - kh * 3;
        // A: whi[khw][m0+row][cb+kg]
        const int aidx = khw * 32768 + cb + arow;
        copy16_g2l(&Ahi[buf][ldsA], whi + aidx);
        copy16_g2l(&Alo[buf][ldsA], wlo + aidx);
        // B: xhi[b][hin][win][cb+kg], zero-padded
        const int hin = h + kh - 1;
        const int win = row + kw - 1;      // per-thread pixel column
        if ((unsigned)hin < 64u && (unsigned)win < 64u) {
            const int bidx = (pixB + hin * 64 + kw - 1) * 128 + cb + brow;
            copy16_g2l(&Bhi[buf][ldsB], xhi + bidx);
            copy16_g2l(&Blo[buf][ldsB], xlo + bidx);
        } else {
            uint4 z = {};
            *(uint4*)&Bhi[buf][ldsB] = z;
            *(uint4*)&Blo[buf][ldsB] = z;
        }
    };

    stage(0, 0);
    wait_async0();
    __syncthreads();

    for (int ks = 0; ks < 36; ++ks) {
        const int cur = ks & 1;
        if (ks + 1 < 36) stage(ks + 1, cur ^ 1);   // overlap copy with WMMA

        Frag a_h, a_l, b0h, b0l, b1h, b1l;
        a_h.q[0] = *(const uint4*)&Ahi[cur][aoff0];
        a_h.q[1] = *(const uint4*)&Ahi[cur][aoff1];
        a_l.q[0] = *(const uint4*)&Alo[cur][aoff0];
        a_l.q[1] = *(const uint4*)&Alo[cur][aoff1];
        b0h.q[0] = *(const uint4*)&Bhi[cur][b0off];
        b0h.q[1] = *(const uint4*)&Bhi[cur][b0off + 8];
        b0l.q[0] = *(const uint4*)&Blo[cur][b0off];
        b0l.q[1] = *(const uint4*)&Blo[cur][b0off + 8];
        b1h.q[0] = *(const uint4*)&Bhi[cur][b1off];
        b1h.q[1] = *(const uint4*)&Bhi[cur][b1off + 8];
        b1l.q[0] = *(const uint4*)&Blo[cur][b1off];
        b1l.q[1] = *(const uint4*)&Blo[cur][b1off + 8];

        acc0 = WMMA_BF16(a_h.v, b0l.v, acc0);
        acc0 = WMMA_BF16(a_l.v, b0h.v, acc0);
        acc0 = WMMA_BF16(a_h.v, b0h.v, acc0);
        acc1 = WMMA_BF16(a_h.v, b1l.v, acc1);
        acc1 = WMMA_BF16(a_l.v, b1h.v, acc1);
        acc1 = WMMA_BF16(a_h.v, b1h.v, acc1);

        wait_async0();       // next-tile async copies landed in LDS
        __syncthreads();     // all waves done reading cur, next buf valid
    }

    // Epilogue: VGPR r -> M=r (lanes 0-15) / r+8 (lanes 16-31); N=l15.
    const int coutBase = m0 + mt * 16 + lh * 8;
    const int wcol     = np * 32 + l15;
    const int obase    = b * 1048576 + h * 64;
#pragma unroll
    for (int r = 0; r < 8; ++r) {
        out[obase + (coutBase + r) * 4096 + wcol]      = acc0[r];
        out[obase + (coutBase + r) * 4096 + wcol + 16] = acc1[r];
    }
}

// ---------------------------------------------------------------------------
// Fallback: self-contained kernel (no workspace), round-1 design.
// ---------------------------------------------------------------------------
__global__ __launch_bounds__(256)
void conv3x3_wmma_bf16x3(const float* __restrict__ x,
                         const float* __restrict__ weight,
                         float* __restrict__ out) {
    __shared__ ushort_t Ahi[64 * 32];
    __shared__ ushort_t Alo[64 * 32];
    __shared__ ushort_t Bhi[64 * 32];
    __shared__ ushort_t Blo[64 * 32];

    const int tid = threadIdx.x;
    const int bx  = blockIdx.x;
    const int m0  = blockIdx.y * 64;
    const int b   = bx >> 6;
    const int h   = bx & 63;

    const int ldrow = tid >> 2;
    const int ldkg  = (tid & 3) * 8;

    const int lane = tid & 31;
    const int wave = tid >> 5;
    const int mt   = wave >> 1;
    const int np   = wave & 1;
    const int l15  = lane & 15;
    const int lh   = lane >> 4;

    const int aoff0 = (mt * 16 + l15) * 32 + lh * 16;
    const int aoff1 = aoff0 + 8;
    const int b0off = (np * 32 + l15) * 32 + lh * 16;
    const int b1off = b0off + 16 * 32;

    f32x8 acc0 = {};
    f32x8 acc1 = {};

    const float* wp0 = weight + (size_t)(m0 + ldrow) * 1152 + ldkg;

    for (int ks = 0; ks < 36; ++ks) {
        const int k0 = ks * 32;
        {
            const float* wp = wp0 + k0;
            float4 w0 = *(const float4*)wp;
            float4 w1 = *(const float4*)(wp + 4);
            float wa[8] = { w0.x, w0.y, w0.z, w0.w, w1.x, w1.y, w1.z, w1.w };
            ushort_t hbuf[8], lbuf[8];
#pragma unroll
            for (int i = 0; i < 8; ++i) split_bf16x2(wa[i], hbuf[i], lbuf[i]);
            *(uint4*)&Ahi[ldrow * 32 + ldkg] = pack8(hbuf);
            *(uint4*)&Alo[ldrow * 32 + ldkg] = pack8(lbuf);
            if (ks + 1 < 36) __builtin_prefetch(wp + 32, 0, 1);
        }
        {
            const int nn = ldrow;
            int kk  = k0 + ldkg;
            unsigned cin = (unsigned)kk / 9u;
            unsigned r   = (unsigned)kk - cin * 9u;
            ushort_t hbuf[8], lbuf[8];
#pragma unroll
            for (int i = 0; i < 8; ++i) {
                int kh  = (int)r / 3;
                int kw  = (int)r - kh * 3;
                int hin = h + kh - 1;
                int win = nn + kw - 1;
                float v = 0.0f;
                if ((unsigned)hin < 64u && (unsigned)win < 64u)
                    v = x[(((b * 128 + (int)cin) * 64 + hin) << 6) + win];
                split_bf16x2(v, hbuf[i], lbuf[i]);
                if (++r == 9u) { r = 0u; ++cin; }
            }
            *(uint4*)&Bhi[nn * 32 + ldkg] = pack8(hbuf);
            *(uint4*)&Blo[nn * 32 + ldkg] = pack8(lbuf);
        }

        __syncthreads();

        Frag a_h, a_l, b0h, b0l, b1h, b1l;
        a_h.q[0] = *(const uint4*)&Ahi[aoff0];  a_h.q[1] = *(const uint4*)&Ahi[aoff1];
        a_l.q[0] = *(const uint4*)&Alo[aoff0];  a_l.q[1] = *(const uint4*)&Alo[aoff1];
        b0h.q[0] = *(const uint4*)&Bhi[b0off];  b0h.q[1] = *(const uint4*)&Bhi[b0off + 8];
        b0l.q[0] = *(const uint4*)&Blo[b0off];  b0l.q[1] = *(const uint4*)&Blo[b0off + 8];
        b1h.q[0] = *(const uint4*)&Bhi[b1off];  b1h.q[1] = *(const uint4*)&Bhi[b1off + 8];
        b1l.q[0] = *(const uint4*)&Blo[b1off];  b1l.q[1] = *(const uint4*)&Blo[b1off + 8];

        acc0 = WMMA_BF16(a_h.v, b0l.v, acc0);
        acc0 = WMMA_BF16(a_l.v, b0h.v, acc0);
        acc0 = WMMA_BF16(a_h.v, b0h.v, acc0);
        acc1 = WMMA_BF16(a_h.v, b1l.v, acc1);
        acc1 = WMMA_BF16(a_l.v, b1h.v, acc1);
        acc1 = WMMA_BF16(a_h.v, b1h.v, acc1);

        __syncthreads();
    }

    const int coutBase = m0 + mt * 16 + lh * 8;
    const int wcol     = np * 32 + l15;
    const int obase    = b * 1048576 + h * 64;
#pragma unroll
    for (int r = 0; r < 8; ++r) {
        out[obase + (coutBase + r) * 4096 + wcol]      = acc0[r];
        out[obase + (coutBase + r) * 4096 + wcol + 16] = acc1[r];
    }
}

extern "C" void kernel_launch(void* const* d_in, const int* in_sizes, int n_in,
                              void* d_out, int out_size, void* d_ws, size_t ws_size,
                              hipStream_t stream) {
    (void)in_sizes; (void)n_in; (void)out_size;
    const float* x      = (const float*)d_in[0];   // [16,128,64,64]
    const float* weight = (const float*)d_in[1];   // [256,128,3,3]
    float*       out    = (float*)d_out;           // [16,256,64,64]

    const size_t W_ELEMS = 9u * 256u * 128u;         // 294912
    const size_t X_ELEMS = 16u * 64u * 64u * 128u;   // 8388608
    const size_t NEED    = 2 * (W_ELEMS + X_ELEMS) * sizeof(unsigned short); // ~34.7 MB

    dim3 grid(16 * 64, 4, 1);
    dim3 block(256, 1, 1);

    if (d_ws != nullptr && ws_size >= NEED) {
        unsigned short* whi = (unsigned short*)d_ws;
        unsigned short* wlo = whi + W_ELEMS;
        unsigned short* xhi = wlo + W_ELEMS;
        unsigned short* xlo = xhi + X_ELEMS;
        prep_weight<<<dim3(1152, 1, 1), block, 0, stream>>>(weight, whi, wlo);
        prep_x<<<dim3(8192, 1, 1), block, 0, stream>>>(x, xhi, xlo);
        conv3x3_wmma_fast<<<grid, block, 0, stream>>>(whi, wlo, xhi, xlo, out);
    } else {
        conv3x3_wmma_bf16x3<<<grid, block, 0, stream>>>(x, weight, out);
    }
}
